// HamiltonianNNModel_19816979103946
// MI455X (gfx1250) — compile-verified, matrix operand order checked by reading, not run on hardware
//
#include <hip/hip_runtime.h>
#include <hip/hip_bf16.h>

// ---------------------------------------------------------------------------
// Hamiltonian-NN gradient: out = (dH/dx) @ J^T for a 32->128->128->1 tanh MLP.
// Strategy: f16 WMMA (v_wmma_f32_16x16x32_f16) for all four matmuls, f32
// accumulate. J is folded into the final B-matrix so the last WMMA's D-tile is
// the final output. Each wave owns a 16-row batch tile; 8 waves per block.
// ---------------------------------------------------------------------------

typedef __attribute__((ext_vector_type(16))) _Float16 v16h;
typedef __attribute__((ext_vector_type(8)))  _Float16 v8h;
typedef __attribute__((ext_vector_type(2)))  _Float16 v2h;
typedef __attribute__((ext_vector_type(8)))  float    v8f;

#define D_IN   32
#define NN     128
#define S0     40     // f16 row stride for [128][32]  (+8 pad, keeps 16B align)
#define S1     136    // f16 row stride for [.][128]   (+8 pad, keeps 16B align)
#define SOUT   36     // f32 row stride for out staging (+4 pad, de-conflict)
#define WAVES  8

__device__ __forceinline__ float fast_tanh(float x) {
#if __has_builtin(__builtin_amdgcn_tanhf)
  return __builtin_amdgcn_tanhf(x);               // gfx1250 v_tanh_f32
#else
  float e = __builtin_amdgcn_exp2f(2.885390082f * x);   // exp(2x)
  return 1.0f - 2.0f * __builtin_amdgcn_rcpf(e + 1.0f);
#endif
}

__device__ __forceinline__ v8f wmma16(v16h a, v16h b, v8f c) {
  // (neg_a, A, neg_b, B, c_mod, C, reuse_a, reuse_b)
  return __builtin_amdgcn_wmma_f32_16x16x32_f16(false, a, false, b,
                                                (short)0, c, false, false);
}

// Load a 16x32-of-K fragment (A or B role) from a row-major f16 LDS array.
// 16-bit WMMA operand layout: lane L (L<16) holds row (row0+L), K = kb..kb+7
// and kb+16..kb+23 with kb=0; lanes 16..31 hold the same rows with kb=8.
__device__ __forceinline__ v16h lds_frag(const _Float16* base, int stride,
                                         int row0, int koff, int lane) {
  int r  = row0 + (lane & 15);
  int kb = koff + ((lane & 16) ? 8 : 0);
  const v8h* q = (const v8h*)(base + r * stride + kb);
  v8h lo = q[0];      // K = kb .. kb+7
  v8h hi = q[2];      // K = kb+16 .. kb+23
  return __builtin_shufflevector(lo, hi, 0, 1, 2, 3, 4, 5, 6, 7,
                                         8, 9, 10, 11, 12, 13, 14, 15);
}

__global__ __launch_bounds__(256, 1)
void hnn_grad_kernel(const float* __restrict__ x,
                     const float* __restrict__ W0,
                     const float* __restrict__ b0,
                     const float* __restrict__ W1,
                     const float* __restrict__ b1,
                     const float* __restrict__ Wout,
                     float* __restrict__ out,
                     int nrows) {
  // B-matrix arrays stored as Bmat[n][k] so B[k][n] = Bmat[n][k].
  __shared__ __align__(16) _Float16 sW0 [NN * S0];      // z0  : W0[n][k]
  __shared__ __align__(16) _Float16 sW0J[D_IN * S1];    // dx  : J-folded W0^T
  __shared__ __align__(16) _Float16 sW1 [NN * S1];      // z1  : W1[n][k]
  __shared__ __align__(16) _Float16 sW1T[NN * S1];      // dh0 : W1[k][n]
  __shared__ float sB0[NN];
  __shared__ float sB1[NN];
  __shared__ float sWo[NN];
  __shared__ __align__(16) _Float16 sStage[WAVES * 16 * S1];   // act transpose
  __shared__ float sOutSt[WAVES * 16 * SOUT];                  // store staging

  const int tid = threadIdx.x;

  // ---- One-time per block: weights -> f16 LDS ----
  for (int i = tid; i < NN * D_IN; i += 256) {
    int n = i >> 5, k = i & 31;
    sW0[n * S0 + k] = (_Float16)W0[i];
  }
  for (int i = tid; i < D_IN * NN; i += 256) {
    int n = i >> 7, k = i & 127;     // n: output col (0..31), k: node (0..127)
    // out[:,n] = (n<16) ? dx_raw[:,n+16] : -dx_raw[:,n-16]; dx_raw uses W0[k][n']
    float v = (n < 16) ? W0[k * D_IN + n + 16] : -W0[k * D_IN + n - 16];
    sW0J[n * S1 + k] = (_Float16)v;
  }
  for (int i = tid; i < NN * NN; i += 256) {
    int n = i >> 7, k = i & 127;
    float v = W1[i];                 // W1[n][k]
    sW1 [n * S1 + k] = (_Float16)v;
    sW1T[k * S1 + n] = (_Float16)v;  // Bmat3[k][n] = W1[n][k]  (transposed)
  }
  for (int i = tid; i < NN; i += 256) {
    sB0[i] = b0[i]; sB1[i] = b1[i]; sWo[i] = Wout[i];
  }
  __syncthreads();

  const int  wave = tid >> 5;
  const int  lane = tid & 31;
  const long tile = (long)blockIdx.x * WAVES + wave;
  const long row0 = tile * 16;
  if (row0 >= nrows) return;        // no barriers after this point

  const int nIdx = lane & 15;       // D-layout: column within n-tile
  const int mo   = (lane & 16) ? 8 : 0;  // D-layout: row offset / A-layout kb

  // ---- A1: load x tile [16 x 32] f32 from global, pack to f16 A-fragment ----
  int arow = (int)row0 + nIdx; if (arow >= nrows) arow = nrows - 1;
  const float* xr = x + (size_t)arow * D_IN + mo;   // mo doubles as kb
  float4 f0 = *(const float4*)(xr + 0);
  float4 f1 = *(const float4*)(xr + 4);
  float4 f2 = *(const float4*)(xr + 16);
  float4 f3 = *(const float4*)(xr + 20);
  v16h a1;
  a1[0]  = (_Float16)f0.x; a1[1]  = (_Float16)f0.y;
  a1[2]  = (_Float16)f0.z; a1[3]  = (_Float16)f0.w;
  a1[4]  = (_Float16)f1.x; a1[5]  = (_Float16)f1.y;
  a1[6]  = (_Float16)f1.z; a1[7]  = (_Float16)f1.w;
  a1[8]  = (_Float16)f2.x; a1[9]  = (_Float16)f2.y;
  a1[10] = (_Float16)f2.z; a1[11] = (_Float16)f2.w;
  a1[12] = (_Float16)f3.x; a1[13] = (_Float16)f3.y;
  a1[14] = (_Float16)f3.z; a1[15] = (_Float16)f3.w;

  _Float16* st = sStage + wave * (16 * S1);

  // ---- Matmul 1: z0 = x @ W0^T  (K=32 in one step), + b0, tanh ----
  v2h dpk[8][4];    // packed (1 - h0^2) kept in D-layout for the backward pass
#pragma unroll
  for (int t = 0; t < 8; ++t) {
    v8f c = {};
    c = wmma16(a1, lds_frag(sW0, S0, t * 16, 0, lane), c);
    float bb = sB0[t * 16 + nIdx];
    float d[8];
#pragma unroll
    for (int r = 0; r < 8; ++r) {
      float h = fast_tanh(c[r] + bb);
      st[(mo + r) * S1 + t * 16 + nIdx] = (_Float16)h;   // D->A via LDS
      d[r] = 1.0f - h * h;
    }
#pragma unroll
    for (int j = 0; j < 4; ++j) {
      v2h p; p[0] = (_Float16)d[2 * j]; p[1] = (_Float16)d[2 * j + 1];
      dpk[t][j] = p;
    }
  }

  // ---- Matmul 2: z1 = h0 @ W1^T  (K=128), + b1, tanh, dz1 = Wout*(1-h1^2) ----
  v16h a2[4];
#pragma unroll
  for (int g = 0; g < 4; ++g) a2[g] = lds_frag(st, S1, 0, g * 32, lane);
  v8f acc[8];
#pragma unroll
  for (int t = 0; t < 8; ++t) {
    v8f c = {};
#pragma unroll
    for (int g = 0; g < 4; ++g)
      c = wmma16(a2[g], lds_frag(sW1, S1, t * 16, g * 32, lane), c);
    acc[t] = c;
  }
#pragma unroll
  for (int t = 0; t < 8; ++t) {
    float bb = sB1[t * 16 + nIdx];
    float wo = sWo[t * 16 + nIdx];
#pragma unroll
    for (int r = 0; r < 8; ++r) {
      float h = fast_tanh(acc[t][r] + bb);
      st[(mo + r) * S1 + t * 16 + nIdx] = (_Float16)(wo * (1.0f - h * h));
    }
  }

  // ---- Matmul 3: dh0 = dz1 @ W1 (via W1^T array), dz0 = dh0*(1-h0^2) ----
  v16h a3[4];
#pragma unroll
  for (int g = 0; g < 4; ++g) a3[g] = lds_frag(st, S1, 0, g * 32, lane);
#pragma unroll
  for (int t = 0; t < 8; ++t) {
    v8f c = {};
#pragma unroll
    for (int g = 0; g < 4; ++g)
      c = wmma16(a3[g], lds_frag(sW1T, S1, t * 16, g * 32, lane), c);
#pragma unroll
    for (int r = 0; r < 8; ++r) {
      float dz0 = c[r] * (float)dpk[t][r >> 1][r & 1];
      st[(mo + r) * S1 + t * 16 + nIdx] = (_Float16)dz0;
    }
  }

  // ---- Matmul 4: out = dz0 @ (J-folded W0)  (N=32) ----
  v16h a4[4];
#pragma unroll
  for (int g = 0; g < 4; ++g) a4[g] = lds_frag(st, S1, 0, g * 32, lane);
  float* os = sOutSt + wave * (16 * SOUT);
#pragma unroll
  for (int t = 0; t < 2; ++t) {
    v8f c = {};
#pragma unroll
    for (int g = 0; g < 4; ++g)
      c = wmma16(a4[g], lds_frag(sW0J, S1, t * 16, g * 32, lane), c);
#pragma unroll
    for (int r = 0; r < 8; ++r)
      os[(mo + r) * SOUT + t * 16 + nIdx] = c[r];
  }

  // ---- Coalesced b128 stores of the 16x32 f32 output tile ----
#pragma unroll
  for (int it = 0; it < 4; ++it) {
    int flat = it * 32 + lane;        // 0..127 float4 slots
    int rr   = flat >> 3;             // row 0..15
    int cc   = (flat & 7) * 4;        // col 0,4,..,28
    long grow = row0 + rr;
    if (grow < nrows) {
      float4 v;
      v.x = os[rr * SOUT + cc + 0];
      v.y = os[rr * SOUT + cc + 1];
      v.z = os[rr * SOUT + cc + 2];
      v.w = os[rr * SOUT + cc + 3];
      *(float4*)(out + grow * D_IN + cc) = v;
    }
  }
}

extern "C" void kernel_launch(void* const* d_in, const int* in_sizes, int n_in,
                              void* d_out, int out_size, void* d_ws, size_t ws_size,
                              hipStream_t stream) {
  // setup_inputs order: t, x, W0, b0, W1, b1, Wout, bout, J
  const float* x    = (const float*)d_in[1];
  const float* W0   = (const float*)d_in[2];
  const float* b0   = (const float*)d_in[3];
  const float* W1   = (const float*)d_in[4];
  const float* b1   = (const float*)d_in[5];
  const float* Wout = (const float*)d_in[6];
  float* out = (float*)d_out;

  int nrows  = in_sizes[1] / D_IN;                 // 500000
  int tiles  = (nrows + 15) / 16;                  // 16-row wave tiles
  int blocks = (tiles + WAVES - 1) / WAVES;

  hipLaunchKernelGGL(hnn_grad_kernel, dim3(blocks), dim3(256), 0, stream,
                     x, W0, b0, W1, b1, Wout, out, nrows);
}
